// DecoderInitWrapper_2473901162607
// MI455X (gfx1250) — matrix operand order repaired and checked by reading, not compiled, not run on hardware
//
#include <hip/hip_runtime.h>

// ---------------------------------------------------------------------------
// MI455X (gfx1250) decoder forward: bf16 WMMA GEMMs (double-buffered LDS with
// GLOBAL_LOAD_ASYNC_TO_LDS for activations) + flash-style WMMA attention.
// B=1 S=2048 V=32000 H=1024 L=4 NH=16 NKV=4 HD=64 I=4096 A=256
// ---------------------------------------------------------------------------

typedef __attribute__((ext_vector_type(16))) __bf16          v16bf;
typedef __attribute__((ext_vector_type(8)))  float           v8f;
typedef __attribute__((ext_vector_type(8)))  unsigned short  ush8;

#define S_   2048
#define H_   1024
#define V_   32000
#define L_   4
#define NH_  16
#define NKV_ 4
#define HD_  64
#define I_   4096
#define A_   256

__device__ __forceinline__ unsigned short f2bf(float f) {
  unsigned int u = __builtin_bit_cast(unsigned int, f);
  u += 0x7FFFu + ((u >> 16) & 1u);           // round-to-nearest-even
  return (unsigned short)(u >> 16);
}
__device__ __forceinline__ unsigned int f2bf_pk(float lo, float hi) {
  return (unsigned int)f2bf(lo) | ((unsigned int)f2bf(hi) << 16);
}

union Frag { v16bf v; ush8 h[2]; };

// A-fragment (16x32 bf16): lane-half holds 16 contiguous K {2*khalf..2*khalf+15}
__device__ __forceinline__ v16bf load_a_frag(const unsigned short* p, int khalf) {
  Frag u;
  u.h[0] = *(const ush8*)(p + 2 * khalf);
  u.h[1] = *(const ush8*)(p + 2 * khalf + 8);
  return u.v;
}
// B-fragment (32x16 bf16): lane = column, 16 contiguous K per lane-half
__device__ __forceinline__ v16bf load_b_frag(const unsigned short* p) {
  Frag u;
  u.h[0] = *(const ush8*)(p);
  u.h[1] = *(const ush8*)(p + 8);
  return u.v;
}
__device__ __forceinline__ v8f wmma_bf16(v16bf a, v16bf b, v8f c) {
  return __builtin_amdgcn_wmma_f32_16x16x32_bf16(false, a, false, b,
                                                 (short)0, c, false, false);
}

// per-lane async DMA: 16 bytes global -> LDS, tracked by ASYNCcnt
__device__ __forceinline__ void async_g2l_b128(void* lds, const void* gaddr) {
  const unsigned lo = (unsigned)(size_t)lds;  // low 32 bits = LDS byte offset
  const unsigned long long ga = (unsigned long long)(size_t)gaddr;
  __asm__ volatile("global_load_async_to_lds_b128 %0, %1, off"
                   :: "v"(lo), "v"(ga) : "memory");
}
__device__ __forceinline__ void wait_async0() {
  __asm__ volatile("s_wait_asynccnt 0" ::: "memory");
}

// ---------------------------------------------------------------------------
// embed gather + audio scatter
// ---------------------------------------------------------------------------
__global__ __launch_bounds__(256)
void embed_scatter(const int* __restrict__ ids, const float* __restrict__ embed,
                   const float* __restrict__ audio, const int* __restrict__ aoff,
                   float* __restrict__ h) {
  const int s = blockIdx.x;
  const int off = aoff[0];
  const float* src = (s >= off && s < off + A_)
                   ? (audio + (size_t)(s - off) * H_)
                   : (embed + (size_t)ids[s] * H_);
  float* dst = h + (size_t)s * H_;
  for (int i = threadIdx.x; i < H_; i += 256) dst[i] = src[i];
}

// ---------------------------------------------------------------------------
// RMSNorm (fp32 in, bf16 out)
// ---------------------------------------------------------------------------
__global__ __launch_bounds__(256)
void rmsnorm_bf16(const float* __restrict__ x, const float* __restrict__ w,
                  unsigned short* __restrict__ out, int H) {
  __shared__ float red[8];
  const int s = blockIdx.x;
  const float* xr = x + (size_t)s * H;
  float ss = 0.f;
  for (int i = threadIdx.x; i < H; i += 256) { float v = xr[i]; ss += v * v; }
  ss += __shfl_xor(ss, 1, 32);  ss += __shfl_xor(ss, 2, 32);
  ss += __shfl_xor(ss, 4, 32);  ss += __shfl_xor(ss, 8, 32);
  ss += __shfl_xor(ss, 16, 32);
  const int lane = threadIdx.x & 31, wave = threadIdx.x >> 5;
  if (lane == 0) red[wave] = ss;
  __syncthreads();
  float tot = 0.f;
#pragma unroll
  for (int i = 0; i < 8; ++i) tot += red[i];
  const float sc = rsqrtf(tot / (float)H + 1e-6f);
  unsigned short* o = out + (size_t)s * H;
  for (int i = threadIdx.x; i < H; i += 256) o[i] = f2bf(xr[i] * sc * w[i]);
}

// ---------------------------------------------------------------------------
// Tiled bf16 WMMA GEMM:  C[M,N] = A[M,K](bf16) * B[K,N](fp32->bf16)
// EPI 0: fp32 store   EPI 1: fp32 store + residual   EPI 2: silu(g)*u -> bf16
// block 256 = 8 waves; wave = 16(M) x 64(N); BM=64 BN=128 BK=32.
// 2-stage pipeline: A tile via GLOBAL_LOAD_ASYNC_TO_LDS_B128 (ASYNCcnt),
// B tile global->regs (fp32) overlapped with WMMA, converted+packed to LDS.
// ---------------------------------------------------------------------------
template <int EPI>
__global__ __launch_bounds__(256)
void gemm_bf16(const unsigned short* __restrict__ A, const float* __restrict__ B,
               const float* __restrict__ B2, const float* __restrict__ resid,
               void* __restrict__ outp, int M, int N, int K) {
  __shared__ alignas(16) unsigned short As[2][64 * 32];
  __shared__ alignas(16) unsigned short Bs[2][128 * 32];
  __shared__ alignas(16) unsigned short B2s[2][(EPI == 2) ? 128 * 32 : 16];

  const int tid  = threadIdx.x;
  const int lane = tid & 31, wave = tid >> 5;
  const int wM = wave >> 1, wN = wave & 1;
  const int ln = lane & 15;
  const int khalf = (lane & 16) ? 8 : 0;
  const int kb16  = (lane & 16) ? 16 : 0;
  const int m0 = blockIdx.y * 64, n0 = blockIdx.x * 128;

  v8f acc[4]  = {};
  v8f acc2[4] = {};

  // A loader: 256 lanes x 16B async = 64x32 bf16 tile
  const int ar = tid >> 2, ac = (tid & 3) * 8;
  // B loader: thread owns k-pair (2*kp, 2*kp+1) x 8 columns (nb..nb+7)
  const int kp = tid >> 4, nb = (tid & 15) * 8;

  float br[16], br2[16];

  auto issueA = [&](int kk, int buf) {
    async_g2l_b128(&As[buf][ar * 32 + ac],
                   &A[(size_t)(m0 + ar) * K + kk + ac]);
  };
  auto loadB = [&](const float* __restrict__ Bp, int kk, float* r) {
    const float* p0 = &Bp[(size_t)(kk + 2 * kp) * N + n0 + nb];
    const float* p1 = p0 + N;
    float4 a0 = *(const float4*)(p0);
    float4 a1 = *(const float4*)(p0 + 4);
    float4 b0 = *(const float4*)(p1);
    float4 b1 = *(const float4*)(p1 + 4);
    r[0] = a0.x; r[1] = a0.y; r[2] = a0.z; r[3] = a0.w;
    r[4] = a1.x; r[5] = a1.y; r[6] = a1.z; r[7] = a1.w;
    r[8]  = b0.x; r[9]  = b0.y; r[10] = b0.z; r[11] = b0.w;
    r[12] = b1.x; r[13] = b1.y; r[14] = b1.z; r[15] = b1.w;
    if (kk + 32 < K) __builtin_prefetch(p0 + (size_t)32 * N, 0, 1);
  };
  auto storeB = [&](unsigned short* Bsm, const float* r) {
#pragma unroll
    for (int i = 0; i < 8; ++i)
      *(unsigned int*)&Bsm[(nb + i) * 32 + 2 * kp] = f2bf_pk(r[i], r[8 + i]);
  };

  // ---- prologue: stage tile 0 into buffer 0
  issueA(0, 0);
  loadB(B, 0, br);
  storeB(Bs[0], br);
  if constexpr (EPI == 2) { loadB(B2, 0, br2); storeB(B2s[0], br2); }
  wait_async0();
  __syncthreads();

  for (int kk = 0; kk < K; kk += 32) {
    const int cur = (kk >> 5) & 1, nxt = cur ^ 1;
    const bool more = (kk + 32) < K;
    if (more) {
      issueA(kk + 32, nxt);
      loadB(B, kk + 32, br);
      if constexpr (EPI == 2) loadB(B2, kk + 32, br2);
    }
    v16bf a = load_a_frag(&As[cur][(wM * 16 + ln) * 32], khalf);
#pragma unroll
    for (int t = 0; t < 4; ++t) {
      const int bn = wN * 64 + t * 16 + ln;
      acc[t] = wmma_bf16(a, load_b_frag(&Bs[cur][bn * 32 + kb16]), acc[t]);
      if constexpr (EPI == 2)
        acc2[t] = wmma_bf16(a, load_b_frag(&B2s[cur][bn * 32 + kb16]), acc2[t]);
    }
    if (more) {
      storeB(Bs[nxt], br);
      if constexpr (EPI == 2) storeB(B2s[nxt], br2);
      wait_async0();
    }
    __syncthreads();
  }

  const int hi8 = (lane & 16) ? 8 : 0;
#pragma unroll
  for (int t = 0; t < 4; ++t) {
#pragma unroll
    for (int e = 0; e < 8; ++e) {
      const size_t row = (size_t)(m0 + wM * 16 + e + hi8);
      const size_t col = (size_t)(n0 + wN * 64 + t * 16 + ln);
      const size_t idx = row * (size_t)N + col;
      if constexpr (EPI == 0) {
        ((float*)outp)[idx] = acc[t][e];
      } else if constexpr (EPI == 1) {
        ((float*)outp)[idx] = acc[t][e] + resid[idx];
      } else {
        const float g = acc[t][e];
        const float u = acc2[t][e];
        ((unsigned short*)outp)[idx] = f2bf((g / (1.0f + __expf(-g))) * u);
      }
    }
  }
}

// ---------------------------------------------------------------------------
// per-(s, head) wave: head RMSNorm (q/k) + RoPE; v pass-through.
// writes bf16 q[NH,S,HD], k[NKV,S,HD], v^T[NKV,HD,S] and fp32 present k/v.
// gridDim = (S, NH + 2*NKV), block = 32
// ---------------------------------------------------------------------------
__global__ __launch_bounds__(32)
void rope_qkv(const float* __restrict__ qf, const float* __restrict__ kf,
              const float* __restrict__ vf, const float* __restrict__ qnw,
              const float* __restrict__ knw, const int* __restrict__ posids,
              unsigned short* __restrict__ qbf, unsigned short* __restrict__ kbf,
              unsigned short* __restrict__ vbft,
              float* __restrict__ outK, float* __restrict__ outV) {
  const int s = blockIdx.x;
  const int hy = blockIdx.y;
  const int t = threadIdx.x;  // 0..31, owns dims t and t+32

  if (hy >= NH_ + NKV_) {     // v heads: no norm, no rope
    const int kvh = hy - NH_ - NKV_;
    const float* src = vf + (size_t)s * (NKV_ * HD_) + kvh * HD_;
    const float x0 = src[t], x1 = src[t + 32];
    vbft[(size_t)kvh * HD_ * S_ + (size_t)t * S_ + s]        = f2bf(x0);
    vbft[(size_t)kvh * HD_ * S_ + (size_t)(t + 32) * S_ + s] = f2bf(x1);
    float* od = outV + ((size_t)kvh * S_ + s) * HD_;
    od[t] = x0; od[t + 32] = x1;
    return;
  }
  const bool isq = hy < NH_;
  const float* src = isq ? (qf + (size_t)s * (NH_ * HD_) + hy * HD_)
                         : (kf + (size_t)s * (NKV_ * HD_) + (hy - NH_) * HD_);
  const float* nw = isq ? qnw : knw;
  float x0 = src[t], x1 = src[t + 32];
  float ss = x0 * x0 + x1 * x1;
  ss += __shfl_xor(ss, 1, 32);  ss += __shfl_xor(ss, 2, 32);
  ss += __shfl_xor(ss, 4, 32);  ss += __shfl_xor(ss, 8, 32);
  ss += __shfl_xor(ss, 16, 32);
  const float sc = rsqrtf(ss * (1.0f / 64.0f) + 1e-6f);
  x0 = x0 * sc * nw[t];
  x1 = x1 * sc * nw[t + 32];
  const float pos = (float)posids[s];
  const float inv = __expf(-((float)t * (1.0f / 32.0f)) * 9.210340371976184f); // theta^-(t/32)
  float sn, cs;
  __sincosf(pos * inv, &sn, &cs);
  const float o0 = x0 * cs - x1 * sn;   // x*cos + rotate_half(x)*sin
  const float o1 = x1 * cs + x0 * sn;
  if (isq) {
    unsigned short* d = qbf + ((size_t)hy * S_ + s) * HD_;
    d[t] = f2bf(o0); d[t + 32] = f2bf(o1);
  } else {
    const int kvh = hy - NH_;
    unsigned short* d = kbf + ((size_t)kvh * S_ + s) * HD_;
    d[t] = f2bf(o0); d[t + 32] = f2bf(o1);
    float* od = outK + ((size_t)kvh * S_ + s) * HD_;
    od[t] = o0; od[t + 32] = o1;
  }
}

// ---------------------------------------------------------------------------
// Flash-style causal attention. One wave per 16-row q tile, 32 keys / iter.
// gridDim = (S/128, NH), block 256 (8 waves). Online softmax with 16-lane
// shfl reductions; probs staged through per-wave LDS into A-frag layout.
// ---------------------------------------------------------------------------
__global__ __launch_bounds__(256)
void attn_kernel(const unsigned short* __restrict__ qbf,
                 const unsigned short* __restrict__ kbf,
                 const unsigned short* __restrict__ vbft,
                 unsigned short* __restrict__ obf) {
  __shared__ alignas(16) unsigned short P[8 * 16 * 32];
  const int lane = threadIdx.x & 31;
  const int wave = threadIdx.x >> 5;
  const int ln = lane & 15;
  const int hi8   = (lane & 16) ? 8 : 0;
  const int khalf = (lane & 16) ? 8 : 0;
  const int kb16  = (lane & 16) ? 16 : 0;
  const int head = blockIdx.y;
  const int kv = head >> 2;                    // N_REP = 4
  const int q0 = (blockIdx.x * 8 + wave) * 16;

  const unsigned short* qrow = qbf + ((size_t)head * S_ + q0 + ln) * HD_;
  const v16bf aq0 = load_a_frag(qrow, khalf);        // hd 0..31
  const v16bf aq1 = load_a_frag(qrow + 32, khalf);   // hd 32..63

  const unsigned short* kbase = kbf + (size_t)kv * S_ * HD_;
  const unsigned short* vbase = vbft + (size_t)kv * HD_ * S_;
  unsigned short* Pw = &P[wave * 16 * 32];

  v8f o0 = {}, o1 = {}, o2 = {}, o3 = {};
  float m[8], lsum[8];
#pragma unroll
  for (int e = 0; e < 8; ++e) { m[e] = -1e30f; lsum[e] = 0.f; }

  for (int k0 = 0; k0 <= q0 + 15; k0 += 32) {
    v8f s0 = {}, s1 = {};
    {
      const unsigned short* kr = kbase + (size_t)(k0 + ln) * HD_ + kb16;
      s0 = wmma_bf16(aq0, load_b_frag(kr), s0);
      s0 = wmma_bf16(aq1, load_b_frag(kr + 32), s0);
    }
    {
      const unsigned short* kr = kbase + (size_t)(k0 + 16 + ln) * HD_ + kb16;
      s1 = wmma_bf16(aq0, load_b_frag(kr), s1);
      s1 = wmma_bf16(aq1, load_b_frag(kr + 32), s1);
    }
#pragma unroll
    for (int e = 0; e < 8; ++e) {
      const int rowg = q0 + e + hi8;
      float v0 = s0[e] * 0.125f;                 // HD^-0.5
      float v1 = s1[e] * 0.125f;
      if (k0 + ln      > rowg) v0 = -1e30f;      // causal mask
      if (k0 + 16 + ln > rowg) v1 = -1e30f;
      float mx = fmaxf(v0, v1);
      mx = fmaxf(mx, __shfl_xor(mx, 1, 32));
      mx = fmaxf(mx, __shfl_xor(mx, 2, 32));
      mx = fmaxf(mx, __shfl_xor(mx, 4, 32));
      mx = fmaxf(mx, __shfl_xor(mx, 8, 32));
      const float mn = fmaxf(m[e], mx);
      const float r = __expf(m[e] - mn);
      m[e] = mn;
      const float p0 = __expf(v0 - mn);
      const float p1 = __expf(v1 - mn);
      float su = p0 + p1;
      su += __shfl_xor(su, 1, 32);
      su += __shfl_xor(su, 2, 32);
      su += __shfl_xor(su, 4, 32);
      su += __shfl_xor(su, 8, 32);
      lsum[e] = lsum[e] * r + su;
      o0[e] *= r; o1[e] *= r; o2[e] *= r; o3[e] *= r;
      Pw[(e + hi8) * 32 + ln]      = f2bf(p0);
      Pw[(e + hi8) * 32 + 16 + ln] = f2bf(p1);
    }
    // intra-wave LDS store->load ordering (no block barrier: waves diverge)
    __asm__ volatile("s_wait_dscnt 0" ::: "memory");
    const v16bf ap = load_a_frag(&Pw[ln * 32], khalf);
    o0 = wmma_bf16(ap, load_b_frag(vbase + (size_t)(0  + ln) * S_ + k0 + kb16), o0);
    o1 = wmma_bf16(ap, load_b_frag(vbase + (size_t)(16 + ln) * S_ + k0 + kb16), o1);
    o2 = wmma_bf16(ap, load_b_frag(vbase + (size_t)(32 + ln) * S_ + k0 + kb16), o2);
    o3 = wmma_bf16(ap, load_b_frag(vbase + (size_t)(48 + ln) * S_ + k0 + kb16), o3);
  }
#pragma unroll
  for (int e = 0; e < 8; ++e) {
    const int row = q0 + e + hi8;
    const float inv = 1.0f / lsum[e];
    unsigned short* orow = obf + (size_t)row * (NH_ * HD_) + head * HD_;
    orow[0  + ln] = f2bf(o0[e] * inv);
    orow[16 + ln] = f2bf(o1[e] * inv);
    orow[32 + ln] = f2bf(o2[e] * inv);
    orow[48 + ln] = f2bf(o3[e] * inv);
  }
}

// ---------------------------------------------------------------------------
// host-side orchestration
// ---------------------------------------------------------------------------
extern "C" void kernel_launch(void* const* d_in, const int* in_sizes, int n_in,
                              void* d_out, int out_size, void* d_ws, size_t ws_size,
                              hipStream_t stream) {
  const int*   ids    = (const int*)  d_in[0];
  const int*   pos    = (const int*)  d_in[1];
  const float* audio  = (const float*)d_in[2];
  const int*   aoffi  = (const int*)  d_in[3];
  const float* embed  = (const float*)d_in[4];
  const float* q_w    = (const float*)d_in[5];
  const float* k_w    = (const float*)d_in[6];
  const float* v_w    = (const float*)d_in[7];
  const float* o_w    = (const float*)d_in[8];
  const float* qn_w   = (const float*)d_in[9];
  const float* kn_w   = (const float*)d_in[10];
  const float* ln1_w  = (const float*)d_in[11];
  const float* ln2_w  = (const float*)d_in[12];
  const float* gate_w = (const float*)d_in[13];
  const float* up_w   = (const float*)d_in[14];
  const float* down_w = (const float*)d_in[15];
  const float* norm_w = (const float*)d_in[16];
  const float* lm_w   = (const float*)d_in[17];
  (void)in_sizes; (void)n_in; (void)out_size; (void)ws_size;

  char* ws = (char*)d_ws;
  size_t off = 0;
  auto alloc = [&](size_t bytes) -> void* {
    void* p = ws + off;
    off += (bytes + 255) & ~(size_t)255;
    return p;
  };
  float*          h      = (float*)         alloc((size_t)S_ * H_ * 4);
  unsigned short* normed = (unsigned short*)alloc((size_t)S_ * H_ * 2);
  float*          qf     = (float*)         alloc((size_t)S_ * NH_ * HD_ * 4);
  float*          kf     = (float*)         alloc((size_t)S_ * NKV_ * HD_ * 4);
  float*          vf     = (float*)         alloc((size_t)S_ * NKV_ * HD_ * 4);
  unsigned short* qbf    = (unsigned short*)alloc((size_t)NH_ * S_ * HD_ * 2);
  unsigned short* kbf    = (unsigned short*)alloc((size_t)NKV_ * S_ * HD_ * 2);
  unsigned short* vbft   = (unsigned short*)alloc((size_t)NKV_ * HD_ * S_ * 2);
  unsigned short* attnbf = (unsigned short*)alloc((size_t)S_ * NH_ * HD_ * 2);
  unsigned short* actbf  = (unsigned short*)alloc((size_t)S_ * I_ * 2);

  float* logits  = (float*)d_out;
  float* keysOut = logits + (size_t)S_ * V_;
  float* valsOut = keysOut + (size_t)L_ * NKV_ * S_ * HD_;

  embed_scatter<<<S_, 256, 0, stream>>>(ids, embed, audio, aoffi, h);

  for (int l = 0; l < L_; ++l) {
    rmsnorm_bf16<<<S_, 256, 0, stream>>>(h, ln1_w + (size_t)l * H_, normed, H_);
    gemm_bf16<0><<<dim3((NH_ * HD_) / 128, S_ / 64), 256, 0, stream>>>(
        normed, q_w + (size_t)l * H_ * NH_ * HD_, nullptr, nullptr, qf,
        S_, NH_ * HD_, H_);
    gemm_bf16<0><<<dim3((NKV_ * HD_) / 128, S_ / 64), 256, 0, stream>>>(
        normed, k_w + (size_t)l * H_ * NKV_ * HD_, nullptr, nullptr, kf,
        S_, NKV_ * HD_, H_);
    gemm_bf16<0><<<dim3((NKV_ * HD_) / 128, S_ / 64), 256, 0, stream>>>(
        normed, v_w + (size_t)l * H_ * NKV_ * HD_, nullptr, nullptr, vf,
        S_, NKV_ * HD_, H_);
    rope_qkv<<<dim3(S_, NH_ + 2 * NKV_), 32, 0, stream>>>(
        qf, kf, vf, qn_w + (size_t)l * HD_, kn_w + (size_t)l * HD_, pos,
        qbf, kbf, vbft,
        keysOut + (size_t)l * NKV_ * S_ * HD_,
        valsOut + (size_t)l * NKV_ * S_ * HD_);
    attn_kernel<<<dim3(S_ / 128, NH_), 256, 0, stream>>>(qbf, kbf, vbft, attnbf);
    gemm_bf16<1><<<dim3(H_ / 128, S_ / 64), 256, 0, stream>>>(
        attnbf, o_w + (size_t)l * NH_ * HD_ * H_, nullptr, h, h,
        S_, H_, NH_ * HD_);
    rmsnorm_bf16<<<S_, 256, 0, stream>>>(h, ln2_w + (size_t)l * H_, normed, H_);
    gemm_bf16<2><<<dim3(I_ / 128, S_ / 64), 256, 0, stream>>>(
        normed, gate_w + (size_t)l * H_ * I_, up_w + (size_t)l * H_ * I_,
        nullptr, actbf, S_, I_, H_);
    gemm_bf16<1><<<dim3(H_ / 128, S_ / 64), 256, 0, stream>>>(
        actbf, down_w + (size_t)l * I_ * H_, nullptr, h, h, S_, H_, I_);
  }

  rmsnorm_bf16<<<S_, 256, 0, stream>>>(h, norm_w, normed, H_);
  gemm_bf16<0><<<dim3(V_ / 128, S_ / 64), 256, 0, stream>>>(
      normed, lm_w, nullptr, nullptr, logits, S_, V_, H_);
}
